// TransitionDown_6322191860208
// MI455X (gfx1250) — matrix operand order repaired and checked by reading, not compiled, not run on hardware
//
#include <hip/hip_runtime.h>
#include <hip/hip_bf16.h>

// ---------------------------------------------------------------------------
// Problem constants (from reference): B=8, N=8192, M=4096, C=128, K=16
// ---------------------------------------------------------------------------
#define PB 8
#define PN 8192
#define PM 4096
#define PC 128
#define PK 16
#define BN_EPS 1e-5f

typedef __attribute__((ext_vector_type(16))) __bf16 v16bf;
typedef __attribute__((ext_vector_type(8)))  float  v8f;

union FragBF {
  uint4 q[2];
  v16bf v;
};

union Pack8 {
  uint4  q;
  __bf16 e[8];
};

__device__ inline unsigned long long shfl_xor_u64(unsigned long long v, int mask) {
  unsigned lo = (unsigned)(v & 0xffffffffull);
  unsigned hi = (unsigned)(v >> 32);
  lo = (unsigned)__shfl_xor((int)lo, mask, 32);
  hi = (unsigned)__shfl_xor((int)hi, mask, 32);
  return ((unsigned long long)hi << 32) | (unsigned long long)lo;
}

// ---------------------------------------------------------------------------
// Weight repack: fp32 -> bf16 in WMMA B-fragment order, with a K-permutation:
//   feat channel k<128  -> W col (k+3)   (grouped points)
//   feat channel 128+d  -> W col d       (rel xyz)
//   feat channel >=131  -> 0             (padding to 160)
// This lets the GEMM1 gather do contiguous 16B loads/stores.
// B fragment (32Kx16N bf16): lane holds N=lane%16; halves h=0..15 map to
// K = chunk*32 + 8*(lane/16) + (h&7) + 16*(h/8).  Stored so each lane reads 2x uint4.
// ---------------------------------------------------------------------------
__global__ __launch_bounds__(256) void prep_weights_kernel(
    const float* __restrict__ W1, const float* __restrict__ W2,
    __bf16* __restrict__ w1b, __bf16* __restrict__ w2b) {
  const int n1 = 16 * 5 * 32 * 16;  // 16 col tiles, 5 K-chunks (K padded ->160)
  const int n2 = 16 * 8 * 32 * 16;  // 16 col tiles, 8 K-chunks (K = 256)
  int e = blockIdx.x * 256 + threadIdx.x;
  if (e < n1) {
    int h = e & 15, lane = (e >> 4) & 31, c = (e >> 9) % 5, j = (e >> 9) / 5;
    int n = j * 16 + (lane & 15);
    int k = c * 32 + ((lane >> 4) << 3) + (h & 7) + ((h >> 3) << 4);
    float v;
    if (k < 128)       v = W1[n * 131 + (k + 3)];  // points channels
    else if (k < 131)  v = W1[n * 131 + (k - 128)];  // rel xyz
    else               v = 0.0f;                     // pad
    w1b[e] = (__bf16)v;
  } else if (e < n1 + n2) {
    int e2 = e - n1;
    int h = e2 & 15, lane = (e2 >> 4) & 31, c = (e2 >> 9) & 7, j = e2 >> 12;
    int n = j * 16 + (lane & 15);
    int k = c * 32 + ((lane >> 4) << 3) + (h & 7) + ((h >> 3) << 4);
    w2b[e2] = (__bf16)W2[n * 256 + k];
  }
}

// ---------------------------------------------------------------------------
// kNN: one wave32 per centroid. Per-lane register top-16 (threshold-guarded
// insertion), then 16 rounds of packed-u64 wave-min merges through LDS.
// Also emits new_pos into d_out.
// ---------------------------------------------------------------------------
__global__ __launch_bounds__(256) void knn_kernel(
    const float* __restrict__ pos, const int* __restrict__ idx,
    int* __restrict__ knn, float* __restrict__ newpos) {
  const int tid = threadIdx.x, lane = tid & 31, wave = tid >> 5;
  const int cm = blockIdx.x * 8 + wave;       // centroid id in [0, B*M)
  const int b = cm >> 12;                     // / 4096

  __shared__ unsigned long long cand[8 * 32 * 16];
  unsigned long long* my = &cand[(wave * 32 + lane) * 16];

  const float* posb = pos + (size_t)b * PN * 3;
  const int ci = idx[cm];
  const float cx = posb[ci * 3 + 0];
  const float cy = posb[ci * 3 + 1];
  const float cz = posb[ci * 3 + 2];
  if (lane == 0) {
    newpos[cm * 3 + 0] = cx;
    newpos[cm * 3 + 1] = cy;
    newpos[cm * 3 + 2] = cz;
  }

  float bd[PK];
  int   bi[PK];
#pragma unroll
  for (int j = 0; j < PK; ++j) { bd[j] = 3.4e38f; bi[j] = 0; }

  for (int n = lane; n < PN; n += 32) {
    float dx = posb[n * 3 + 0] - cx;
    float dy = posb[n * 3 + 1] - cy;
    float dz = posb[n * 3 + 2] - cz;
    float d = fmaf(dx, dx, fmaf(dy, dy, dz * dz));
    if (d < bd[PK - 1]) {                // threshold: most candidates skip
      float key = d; int ki = n;
#pragma unroll
      for (int j = 0; j < PK; ++j) {     // register insertion (no dyn index)
        bool cl = key < bd[j];
        float tb = bd[j]; int ti = bi[j];
        bd[j] = cl ? key : tb; bi[j] = cl ? ki : ti;
        key = cl ? tb : key;  ki = cl ? ti : ki;
      }
    }
  }

#pragma unroll
  for (int j = 0; j < PK; ++j)
    my[j] = ((unsigned long long)__float_as_uint(bd[j]) << 32) |
            ((unsigned long long)((unsigned)bi[j]) << 5) | (unsigned)lane;

  // 16 rounds: wave-min over 32 sorted heads (dist>=0 so float-bit order ok)
  int head = 0;
  for (int t = 0; t < PK; ++t) {
    unsigned long long p = (head < PK) ? my[head] : ~0ull;
#pragma unroll
    for (int mask = 16; mask >= 1; mask >>= 1) {
      unsigned long long o = shfl_xor_u64(p, mask);
      p = (o < p) ? o : p;
    }
    if (lane == 0) knn[cm * PK + t] = (int)((p >> 5) & 8191ull);
    if ((int)(p & 31ull) == lane) head++;
  }
}

// ---------------------------------------------------------------------------
// GEMM1: gather feat rows (pts(0..127), rel(128..130), pad->160) into LDS as
// bf16 via 2x b128 loads + b128 LDS stores, then 16x16x32 bf16 WMMA, f32 acc.
// One tile = one (b,m) centroid (16 rows = K neighbors).  8 waves x 2 column
// tiles = 256 output channels.  BN1 sum/sumsq partials in LDS; one global
// atomic per channel per block at kernel end.
// ---------------------------------------------------------------------------
__global__ __launch_bounds__(256) void gemm1_kernel(
    const float* __restrict__ pos, const float* __restrict__ points,
    const int* __restrict__ knn, const float* __restrict__ newpos,
    const __bf16* __restrict__ w1b, const float* __restrict__ b1,
    __bf16* __restrict__ h1, float* __restrict__ stats1, int numTiles) {
  __shared__ __align__(16) __bf16 As[16 * 160];
  __shared__ float lsum[256], lsq[256];
  const int tid = threadIdx.x, lane = tid & 31, wave = tid >> 5;
  lsum[tid] = 0.0f; lsq[tid] = 0.0f;
  __syncthreads();

  for (int t = blockIdx.x; t < numTiles; t += gridDim.x) {
    const int b = t >> 12;
    {  // ---- stage A tile: vectorized gather + bf16 convert ----
      int r = tid >> 4, c0 = tid & 15;
      int nidx = knn[t * PK + r];
      const float* prow = points + ((size_t)b * PN + nidx) * PC;
      // 8 contiguous point channels per thread: 2x float4 (16B) loads
      float4 p0 = *(const float4*)(prow + 8 * c0);
      float4 p1 = *(const float4*)(prow + 8 * c0 + 4);
      Pack8 o;
      o.e[0] = (__bf16)p0.x; o.e[1] = (__bf16)p0.y;
      o.e[2] = (__bf16)p0.z; o.e[3] = (__bf16)p0.w;
      o.e[4] = (__bf16)p1.x; o.e[5] = (__bf16)p1.y;
      o.e[6] = (__bf16)p1.z; o.e[7] = (__bf16)p1.w;
      *(uint4*)&As[r * 160 + 8 * c0] = o.q;           // channels 8c0..8c0+7
      if (c0 < 3) {                                   // rel xyz at 128..130
        const float* gp = pos + ((size_t)b * PN + nidx) * 3;
        float cc = newpos[t * 3 + c0];
        As[r * 160 + 128 + c0] = (__bf16)(gp[c0] - cc);
      }
      for (int c = 131 + c0; c < 160; c += 16)        // zero padding
        As[r * 160 + c] = (__bf16)0.0f;
    }
    __syncthreads();

    v8f accs[2] = {};
    const int j0 = wave * 2;
#pragma unroll
    for (int c = 0; c < 5; ++c) {
      FragBF a;
      const int row = lane & 15;
      const int kb = c * 32 + ((lane >> 4) << 3);  // A: K = kb..kb+7 / +16..+23
      a.q[0] = *(const uint4*)&As[row * 160 + kb];
      a.q[1] = *(const uint4*)&As[row * 160 + kb + 16];
#pragma unroll
      for (int jj = 0; jj < 2; ++jj) {
        FragBF bf;
        const __bf16* bp = w1b + ((size_t)(((j0 + jj) * 5 + c) * 32 + lane)) * 16;
        bf.q[0] = *(const uint4*)bp;
        bf.q[1] = *(const uint4*)(bp + 8);
        accs[jj] = __builtin_amdgcn_wmma_f32_16x16x32_bf16(
            false, a.v, false, bf.v, (short)0, accs[jj], false, false);
      }
    }

    // ---- epilogue: bias, store bf16, BN stats partials ----
#pragma unroll
    for (int jj = 0; jj < 2; ++jj) {
      const int col = (j0 + jj) * 16 + (lane & 15);
      const float bias = b1[col];
      const int rbase = (lane >> 4) << 3;  // D rows: lanes0-15 M=v, 16-31 M=8+v
      float s = 0.0f, s2 = 0.0f;
#pragma unroll
      for (int v = 0; v < 8; ++v) {
        float val = accs[jj][v] + bias;
        h1[((size_t)t * 16 + rbase + v) * 256 + col] = (__bf16)val;
        s += val; s2 = fmaf(val, val, s2);
      }
      s += __shfl_xor(s, 16, 32);
      s2 += __shfl_xor(s2, 16, 32);
      if (lane < 16) { lsum[col] += s; lsq[col] += s2; }  // unique col per thread
    }
    __syncthreads();
  }
  atomicAdd(&stats1[tid], lsum[tid]);
  atomicAdd(&stats1[256 + tid], lsq[tid]);
}

// ---------------------------------------------------------------------------
// BN finalize: stats -> per-channel scale a = g*rsqrt(var+eps), shift
// c = beta - mean*a  (BN folded into affine).
// ---------------------------------------------------------------------------
__global__ __launch_bounds__(256) void finalize_bn_kernel(
    const float* __restrict__ stats, const float* __restrict__ gamma,
    const float* __restrict__ beta, float* __restrict__ bn, float invR) {
  int c = threadIdx.x;
  float mean = stats[c] * invR;
  float var = fmaf(-mean, mean, stats[256 + c] * invR);
  float a = gamma[c] * rsqrtf(var + BN_EPS);
  bn[c] = a;
  bn[256 + c] = fmaf(-mean, a, beta[c]);
}

// ---------------------------------------------------------------------------
// GEMM2: h1(bf16) -> BN1 affine + ReLU on the fly -> WMMA (256x256) -> bias,
// BN2 stats, and K-max-pool (wave-local: 8 rows per lane + shfl_xor(16)).
// Pooled pre-BN2 values go to ws (BN2+ReLU are monotone per channel).
// ---------------------------------------------------------------------------
__global__ __launch_bounds__(256) void gemm2_kernel(
    const __bf16* __restrict__ h1, const __bf16* __restrict__ w2b,
    const float* __restrict__ bn1, const float* __restrict__ b2,
    float* __restrict__ pooled, float* __restrict__ stats2, int numTiles) {
  __shared__ __align__(16) __bf16 As[16 * 256];
  __shared__ float lsum[256], lsq[256];
  __shared__ float sbn[512];
  const int tid = threadIdx.x, lane = tid & 31, wave = tid >> 5;
  lsum[tid] = 0.0f; lsq[tid] = 0.0f;
  sbn[tid] = bn1[tid];
  sbn[256 + tid] = bn1[256 + tid];
  __syncthreads();

  for (int t = blockIdx.x; t < numTiles; t += gridDim.x) {
    {  // ---- stage: h1 -> relu(a1*x + c1) -> bf16, 16 channels/thread ----
      int r = tid >> 4, c0 = tid & 15;
      size_t rb = ((size_t)t * 16 + r) * 256 + 16 * c0;
      Pack8 i0, i1, o0, o1;
      i0.q = *(const uint4*)(h1 + rb);
      i1.q = *(const uint4*)(h1 + rb + 8);
#pragma unroll
      for (int i = 0; i < 8; ++i) {
        int ch = 16 * c0 + i;
        o0.e[i] = (__bf16)fmaxf(fmaf(sbn[ch], (float)i0.e[i], sbn[256 + ch]), 0.0f);
        o1.e[i] = (__bf16)fmaxf(
            fmaf(sbn[ch + 8], (float)i1.e[i], sbn[256 + ch + 8]), 0.0f);
      }
      *(uint4*)&As[r * 256 + 16 * c0] = o0.q;
      *(uint4*)&As[r * 256 + 16 * c0 + 8] = o1.q;
    }
    __syncthreads();

    v8f accs[2] = {};
    const int j0 = wave * 2;
#pragma unroll
    for (int c = 0; c < 8; ++c) {
      FragBF a;
      const int row = lane & 15;
      const int kb = c * 32 + ((lane >> 4) << 3);
      a.q[0] = *(const uint4*)&As[row * 256 + kb];
      a.q[1] = *(const uint4*)&As[row * 256 + kb + 16];
#pragma unroll
      for (int jj = 0; jj < 2; ++jj) {
        FragBF bf;
        const __bf16* bp = w2b + ((size_t)(((j0 + jj) * 8 + c) * 32 + lane)) * 16;
        bf.q[0] = *(const uint4*)bp;
        bf.q[1] = *(const uint4*)(bp + 8);
        accs[jj] = __builtin_amdgcn_wmma_f32_16x16x32_bf16(
            false, a.v, false, bf.v, (short)0, accs[jj], false, false);
      }
    }

#pragma unroll
    for (int jj = 0; jj < 2; ++jj) {
      const int col = (j0 + jj) * 16 + (lane & 15);
      const float bias = b2[col];
      float s = 0.0f, s2 = 0.0f, mx = -3.4e38f;
#pragma unroll
      for (int v = 0; v < 8; ++v) {
        float val = accs[jj][v] + bias;
        s += val; s2 = fmaf(val, val, s2);
        mx = fmaxf(mx, val);
      }
      s += __shfl_xor(s, 16, 32);
      s2 += __shfl_xor(s2, 16, 32);
      mx = fmaxf(mx, __shfl_xor(mx, 16, 32));   // max over all 16 neighbors
      if (lane < 16) {
        lsum[col] += s; lsq[col] += s2;
        pooled[(size_t)t * 256 + col] = mx;
      }
    }
    __syncthreads();
  }
  atomicAdd(&stats2[tid], lsum[tid]);
  atomicAdd(&stats2[256 + tid], lsq[tid]);
}

// ---------------------------------------------------------------------------
// Output: BN2 affine + ReLU on pooled tensor -> d_out (after new_pos).
// ---------------------------------------------------------------------------
__global__ __launch_bounds__(256) void out_kernel(
    const float* __restrict__ pooled, const float* __restrict__ bn2,
    float* __restrict__ outp, int total) {
  int i = blockIdx.x * 256 + threadIdx.x;
  if (i < total) {
    int c = i & 255;
    outp[i] = fmaxf(fmaf(bn2[c], pooled[i], bn2[256 + c]), 0.0f);
  }
}

// ---------------------------------------------------------------------------
extern "C" void kernel_launch(void* const* d_in, const int* in_sizes, int n_in,
                              void* d_out, int out_size, void* d_ws, size_t ws_size,
                              hipStream_t stream) {
  (void)in_sizes; (void)n_in; (void)out_size; (void)ws_size;
  const float* pos    = (const float*)d_in[0];
  const float* points = (const float*)d_in[1];
  const int*   idx    = (const int*)d_in[2];
  const float* W1     = (const float*)d_in[3];
  const float* b1     = (const float*)d_in[4];
  const float* g1     = (const float*)d_in[5];
  const float* be1    = (const float*)d_in[6];
  const float* W2     = (const float*)d_in[7];
  const float* b2     = (const float*)d_in[8];
  const float* g2     = (const float*)d_in[9];
  const float* be2    = (const float*)d_in[10];

  float* outBase = (float*)d_out;
  float* newpos  = outBase;                            // [B,M,3]
  float* outpts  = outBase + (size_t)PB * PM * 3;      // [B,M,256]

  // workspace layout
  char* ws = (char*)d_ws;
  size_t off = 0;
  int* knn = (int*)(ws + off);          off += (size_t)PB * PM * PK * 4;   // 2 MB
  __bf16* w1b = (__bf16*)(ws + off);    off += (size_t)16 * 5 * 32 * 16 * 2;
  __bf16* w2b = (__bf16*)(ws + off);    off += (size_t)16 * 8 * 32 * 16 * 2;
  off = (off + 255) & ~(size_t)255;
  float* stats = (float*)(ws + off);    off += 1024 * 4;   // stats1[512] stats2[512]
  float* bn1 = (float*)(ws + off);      off += 512 * 4;
  float* bn2 = (float*)(ws + off);      off += 512 * 4;
  off = (off + 255) & ~(size_t)255;
  __bf16* h1 = (__bf16*)(ws + off);     off += (size_t)PB * PM * PK * 256 * 2; // 268 MB
  float* pooled = (float*)(ws + off);   off += (size_t)PB * PM * 256 * 4;      // 33 MB

  const int numTiles = PB * PM;                 // 32768 (b,m) tiles
  const float invR = 1.0f / (float)((size_t)PB * PM * PK);

  hipMemsetAsync(stats, 0, 1024 * sizeof(float), stream);

  prep_weights_kernel<<<(40960 + 65536 + 255) / 256, 256, 0, stream>>>(W1, W2, w1b, w2b);
  knn_kernel<<<numTiles / 8, 256, 0, stream>>>(pos, idx, knn, newpos);
  gemm1_kernel<<<2048, 256, 0, stream>>>(pos, points, knn, newpos, w1b, b1, h1,
                                         stats, numTiles);
  finalize_bn_kernel<<<1, 256, 0, stream>>>(stats, g1, be1, bn1, invR);
  gemm2_kernel<<<2048, 256, 0, stream>>>(h1, w2b, bn1, b2, pooled, stats + 512,
                                         numTiles);
  finalize_bn_kernel<<<1, 256, 0, stream>>>(stats + 512, g2, be2, bn2, invR);
  out_kernel<<<(PB * PM * 256 + 255) / 256, 256, 0, stream>>>(
      pooled, bn2, outpts, PB * PM * 256);
}
